// Attention_3762391351510
// MI455X (gfx1250) — matrix operand order repaired and checked by reading, not compiled, not run on hardware
//
#include <hip/hip_runtime.h>
#include <hip/hip_bf16.h>
#include <stdint.h>

// Problem constants (from reference)
#define DIMX   512
#define HEADS  8
#define DHEAD  64
#define INNER  512          // HEADS*DHEAD
#define MAXPOS 512
#define BATCH  8
#define SEQ    1024
#define BN     (BATCH*SEQ)  // 8192
#define RELROWS 1040        // 2*MAXPOS+1 = 1025, padded to multiple of 16 (pad rows = 0)
#define RELTILES (RELROWS/16)

typedef __attribute__((ext_vector_type(16))) __bf16    v16bf;
typedef __attribute__((ext_vector_type(8)))  float     v8f;
typedef __attribute__((ext_vector_type(4)))  unsigned int v4u;

union Frag { v16bf h; v4u u[2]; };

static __device__ inline v8f wmma_bf16(const Frag& a, const Frag& b, v8f c) {
  // D = A(16x32 bf16) * B(32x16 bf16) + C(16x16 f32)
  return __builtin_amdgcn_wmma_f32_16x16x32_bf16(false, a.h, false, b.h,
                                                 (short)0, c, false, false);
}

// ---------------- conversion kernels ----------------

__global__ void cvt_f32_bf16(const float* __restrict__ src, __bf16* __restrict__ dst, int n) {
  int i = blockIdx.x * blockDim.x + threadIdx.x;
  if (i < n) dst[i] = (__bf16)src[i];
}

// wT[c][k] = (c<512 ? Wq[k][c] : Wkv[k][c-512]), c in [0,1536), k in [0,512)
__global__ void cvt_wqkvT(const float* __restrict__ Wq, const float* __restrict__ Wkv,
                          __bf16* __restrict__ wT) {
  int i = blockIdx.x * blockDim.x + threadIdx.x;
  if (i >= 1536 * 512) return;
  int kk = i & 511, c = i >> 9;
  float v = (c < INNER) ? Wq[(size_t)kk * INNER + c]
                        : Wkv[(size_t)kk * (2 * INNER) + (c - INNER)];
  wT[i] = (__bf16)v;
}

// woT[c][k] = Wo[k][c]
__global__ void cvt_woT(const float* __restrict__ Wo, __bf16* __restrict__ wT) {
  int i = blockIdx.x * blockDim.x + threadIdx.x;
  if (i >= 512 * 512) return;
  int kk = i & 511, c = i >> 9;
  wT[i] = (__bf16)Wo[(size_t)kk * DIMX + c];
}

// relbf[t][d], t in [0,1040): rows >= 1025 are zero padding (never gathered, safe to read)
__global__ void cvt_rel(const float* __restrict__ rel, __bf16* __restrict__ relbf) {
  int i = blockIdx.x * blockDim.x + threadIdx.x;
  if (i >= RELROWS * DHEAD) return;
  int d = i & 63, t = i >> 6;
  relbf[i] = (__bf16)((t < 2 * MAXPOS + 1) ? rel[(size_t)t * DHEAD + d] : 0.0f);
}

// ---------------- QKV projection GEMM ----------------
// One wave computes a 16x64 tile of [8192 x 1536] = x @ [Wq|Wk|Wv] (4 accumulators,
// A-fragment reused 4x per K-slab). c0 is 64-aligned, so a block's 64 columns fall
// in exactly one of {q,k,v} and one head. q,k stored [bh][n][64]; v as [bh][64][n].
__global__ __launch_bounds__(32) void qkv_gemm(const __bf16* __restrict__ xbf,
                                               const __bf16* __restrict__ wT,
                                               __bf16* __restrict__ qbf,
                                               __bf16* __restrict__ kbf,
                                               __bf16* __restrict__ vT) {
  int lane  = threadIdx.x;
  int nlane = lane & 15, hi = lane >> 4;
  int c0 = blockIdx.x * 64;
  int r0 = blockIdx.y * 16;
  const __bf16* arow = xbf + (size_t)(r0 + nlane) * DIMX;
  const __bf16* brow0 = wT + (size_t)(c0 + nlane) * DIMX;
  v8f acc[4] = {{}, {}, {}, {}};
  for (int kb = 0; kb < DIMX; kb += 32) {
    Frag a;
    a.u[0] = *reinterpret_cast<const v4u*>(arow + kb + hi * 16);
    a.u[1] = *reinterpret_cast<const v4u*>(arow + kb + hi * 16 + 8);
    __builtin_prefetch(arow + kb + 32, 0, 1);
#pragma unroll
    for (int nt = 0; nt < 4; ++nt) {
      const __bf16* brow = brow0 + (size_t)nt * 16 * DIMX;
      Frag b;
      b.u[0] = *reinterpret_cast<const v4u*>(brow + kb + hi * 16);
      b.u[1] = *reinterpret_cast<const v4u*>(brow + kb + hi * 16 + 8);
      acc[nt] = wmma_bf16(a, b, acc[nt]);
    }
  }
  int region = c0 >> 9;            // 0 = q, 1 = k, 2 = v (block never straddles)
  int h = (c0 & 511) >> 6;         // head, constant per block
#pragma unroll
  for (int nt = 0; nt < 4; ++nt) {
    for (int r = 0; r < 8; ++r) {
      int R  = r0 + r + hi * 8;
      int bb = R >> 10, nn = R & (SEQ - 1);
      int dd = nt * 16 + nlane;    // within-head dim, 0..63
      __bf16 val = (__bf16)acc[nt][r];
      size_t bhh = (size_t)bb * HEADS + h;
      if (region == 0)      qbf[(bhh * SEQ + nn) * DHEAD + dd] = val;
      else if (region == 1) kbf[(bhh * SEQ + nn) * DHEAD + dd] = val;
      else                  vT [(bhh * DHEAD + dd) * SEQ + nn] = val;
    }
  }
}

// ---------------- fused attention kernel ----------------
// One 4-wave workgroup per (bh, 16-query-row tile); waves cooperate on shared LDS:
// G[16][1040] f32, S[16][1024] f32, invs[16]; P[16][1024] bf16 reuses dead G space.
// 132 KB LDS / 4 waves -> 2 blocks (8 waves) per 320 KB WGP, filling all 4 SIMD32s.
__global__ __launch_bounds__(128) void attn_kernel(const __bf16* __restrict__ qbf,
                                                   const __bf16* __restrict__ kbf,
                                                   const __bf16* __restrict__ vT,
                                                   const __bf16* __restrict__ relbf,
                                                   __bf16* __restrict__ attnbf) {
  extern __shared__ __align__(16) char smem_raw[];
  float*  G    = (float*)smem_raw;          // 16*1040 f32
  float*  S    = G + 16 * RELROWS;          // 16*1024 f32
  float*  invs = S + 16 * SEQ;              // 16 f32
  __bf16* P    = (__bf16*)smem_raw;         // 16*1024 bf16, reuses G space

  int wid   = threadIdx.x >> 5;             // wave 0..3
  int lane  = threadIdx.x & 31;
  int nlane = lane & 15, hi = lane >> 4;
  int q0 = blockIdx.x * 16;                 // query tile base
  int bh = blockIdx.y;                      // 0..63

  const __bf16* Qb = qbf + (size_t)bh * SEQ * DHEAD;
  const __bf16* Kb = kbf + (size_t)bh * SEQ * DHEAD;
  const __bf16* Vt = vT  + (size_t)bh * DHEAD * SEQ;

  // Every wave holds the Q tile as two A-fragments (d = 0..31, 32..63)
  Frag aq[2];
  {
    const __bf16* qrow = Qb + (size_t)(q0 + nlane) * DHEAD;
    for (int s = 0; s < 2; ++s) {
      aq[s].u[0] = *reinterpret_cast<const v4u*>(qrow + s * 32 + hi * 16);
      aq[s].u[1] = *reinterpret_cast<const v4u*>(qrow + s * 32 + hi * 16 + 8);
    }
  }

  // G = Q_tile @ rel_emb^T (pos_attn as WMMA GEMM); t-tiles split across waves
  for (int tt = wid; tt < RELTILES; tt += 4) {
    v8f g = {};
    const __bf16* rrow = relbf + (size_t)(tt * 16 + nlane) * DHEAD;
    for (int s = 0; s < 2; ++s) {
      Frag b;
      b.u[0] = *reinterpret_cast<const v4u*>(rrow + s * 32 + hi * 16);
      b.u[1] = *reinterpret_cast<const v4u*>(rrow + s * 32 + hi * 16 + 8);
      g = wmma_bf16(aq[s], b, g);
    }
    for (int r = 0; r < 8; ++r)
      G[(r + hi * 8) * RELROWS + tt * 16 + nlane] = g[r];
  }
  __syncthreads();

  // S = (Q K^T + gather(G)) * 1/sqrt(64); key tiles split across waves
  for (int kt = wid; kt < SEQ / 16; kt += 4) {
    v8f sf = {};
    const __bf16* krow = Kb + (size_t)(kt * 16 + nlane) * DHEAD;
    for (int s = 0; s < 2; ++s) {
      Frag b;
      b.u[0] = *reinterpret_cast<const v4u*>(krow + s * 32 + hi * 16);
      b.u[1] = *reinterpret_cast<const v4u*>(krow + s * 32 + hi * 16 + 8);
      sf = wmma_bf16(aq[s], b, sf);
    }
    int kr = kt * 16 + nlane;
    for (int r = 0; r < 8; ++r) {
      int m  = r + hi * 8;
      int dl = (q0 + m) - kr;
      dl = dl < -MAXPOS ? -MAXPOS : (dl > MAXPOS ? MAXPOS : dl);
      S[m * SEQ + kr] = (sf[r] + G[m * RELROWS + dl + MAXPOS]) * 0.125f;
    }
  }
  __syncthreads();

  // Row softmax: 4 rows per wave, wave32 shfl reductions; P bf16 into dead G space
  for (int m = wid; m < 16; m += 4) {
    float mx = -3.4e38f;
    for (int j = lane; j < SEQ; j += 32) mx = fmaxf(mx, S[m * SEQ + j]);
    for (int off = 16; off > 0; off >>= 1) mx = fmaxf(mx, __shfl_xor(mx, off, 32));
    float sum = 0.0f;
    for (int j = lane; j < SEQ; j += 32) {
      float p = __expf(S[m * SEQ + j] - mx);
      sum += p;
      P[m * SEQ + j] = (__bf16)p;
    }
    for (int off = 16; off > 0; off >>= 1) sum += __shfl_xor(sum, off, 32);
    if (lane == 0) invs[m] = 1.0f / sum;
  }
  __syncthreads();

  // O = P @ V: one 16-wide d-tile per wave; B-operand from V^T (contiguous per lane)
  int b_ = bh >> 3, h_ = bh & 7;
  const __bf16* prow = P + (size_t)nlane * SEQ;
  {
    int nt = wid;
    v8f o = {};
    const __bf16* vrow = Vt + (size_t)(nt * 16 + nlane) * SEQ;
    for (int kb = 0; kb < SEQ; kb += 32) {
      Frag a, b;
      a.u[0] = *reinterpret_cast<const v4u*>(prow + kb + hi * 16);
      a.u[1] = *reinterpret_cast<const v4u*>(prow + kb + hi * 16 + 8);
      b.u[0] = *reinterpret_cast<const v4u*>(vrow + kb + hi * 16);
      b.u[1] = *reinterpret_cast<const v4u*>(vrow + kb + hi * 16 + 8);
      o = wmma_bf16(a, b, o);
    }
    for (int r = 0; r < 8; ++r) {
      int m = r + hi * 8;
      size_t R = (size_t)b_ * SEQ + q0 + m;
      attnbf[R * INNER + h_ * DHEAD + nt * 16 + nlane] = (__bf16)(o[r] * invs[m]);
    }
  }
}

// ---------------- output projection + bias ----------------
// One wave computes a 16x64 tile of out[8192x512] (4 accumulators, A reuse 4x).
__global__ __launch_bounds__(32) void out_gemm(const __bf16* __restrict__ attnbf,
                                               const __bf16* __restrict__ woT,
                                               const float* __restrict__ bo,
                                               float* __restrict__ out) {
  int lane  = threadIdx.x;
  int nlane = lane & 15, hi = lane >> 4;
  int c0 = blockIdx.x * 64;
  int r0 = blockIdx.y * 16;
  const __bf16* arow  = attnbf + (size_t)(r0 + nlane) * INNER;
  const __bf16* brow0 = woT    + (size_t)(c0 + nlane) * INNER;
  v8f acc[4] = {{}, {}, {}, {}};
  for (int kb = 0; kb < INNER; kb += 32) {
    Frag a;
    a.u[0] = *reinterpret_cast<const v4u*>(arow + kb + hi * 16);
    a.u[1] = *reinterpret_cast<const v4u*>(arow + kb + hi * 16 + 8);
    __builtin_prefetch(arow + kb + 32, 0, 1);
#pragma unroll
    for (int nt = 0; nt < 4; ++nt) {
      const __bf16* brow = brow0 + (size_t)nt * 16 * INNER;
      Frag b;
      b.u[0] = *reinterpret_cast<const v4u*>(brow + kb + hi * 16);
      b.u[1] = *reinterpret_cast<const v4u*>(brow + kb + hi * 16 + 8);
      acc[nt] = wmma_bf16(a, b, acc[nt]);
    }
  }
#pragma unroll
  for (int nt = 0; nt < 4; ++nt) {
    int col = c0 + nt * 16 + nlane;
    float bias = bo[col];
    for (int r = 0; r < 8; ++r) {
      int R = r0 + r + hi * 8;
      out[(size_t)R * DIMX + col] = acc[nt][r] + bias;
    }
  }
}

// ---------------- launcher ----------------
extern "C" void kernel_launch(void* const* d_in, const int* in_sizes, int n_in,
                              void* d_out, int out_size, void* d_ws, size_t ws_size,
                              hipStream_t stream) {
  (void)in_sizes; (void)n_in; (void)out_size; (void)ws_size;
  const float* x   = (const float*)d_in[0];
  const float* Wq  = (const float*)d_in[1];
  const float* Wkv = (const float*)d_in[2];
  const float* rel = (const float*)d_in[3];
  const float* Wo  = (const float*)d_in[4];
  const float* bo  = (const float*)d_in[5];
  float* out = (float*)d_out;

  // Workspace layout (bytes); total ~44.2 MB
  char* ws = (char*)d_ws;
  __bf16* xbf    = (__bf16*)(ws + 0);
  __bf16* qbf    = (__bf16*)(ws + 8388608);
  __bf16* kbf    = (__bf16*)(ws + 16777216);
  __bf16* vTbf   = (__bf16*)(ws + 25165824);
  __bf16* attnbf = (__bf16*)(ws + 33554432);
  __bf16* wqkvT  = (__bf16*)(ws + 41943040);
  __bf16* woT    = (__bf16*)(ws + 43515904);
  __bf16* relbf  = (__bf16*)(ws + 44040192);

  cvt_f32_bf16<<<(BN * DIMX + 255) / 256, 256, 0, stream>>>(x, xbf, BN * DIMX);
  cvt_wqkvT   <<<(1536 * 512 + 255) / 256, 256, 0, stream>>>(Wq, Wkv, wqkvT);
  cvt_woT     <<<(512 * 512 + 255) / 256, 256, 0, stream>>>(Wo, woT);
  cvt_rel     <<<(RELROWS * DHEAD + 255) / 256, 256, 0, stream>>>(rel, relbf);

  qkv_gemm<<<dim3(1536 / 64, BN / 16), 32, 0, stream>>>(xbf, wqkvT, qbf, kbf, vTbf);

  size_t smem = (size_t)(16 * RELROWS + 16 * SEQ + 16) * sizeof(float); // 132,160 B
  attn_kernel<<<dim3(SEQ / 16, BATCH * HEADS), 128, smem, stream>>>(qbf, kbf, vTbf, relbf, attnbf);

  out_gemm<<<dim3(DIMX / 64, BN / 16), 32, 0, stream>>>(attnbf, woT, bo, out);
}
